// ResGNN_20109036880395
// MI455X (gfx1250) — compile-verified, matrix operand order checked by reading, not run on hardware
//
#include <hip/hip_runtime.h>
#include <hip/hip_bf16.h>

#define USER_N 100000
#define ITEM_N 60000
#define NTOT   160000
#define DIM    128
#define EPS_BN 1e-5f

typedef __attribute__((ext_vector_type(16))) _Float16 v16h;
typedef __attribute__((ext_vector_type(8)))  float    v8f;

// out layout: 8 slabs of NTOT*DIM floats: [emb, l1, l2, l3, emb, g1, g2, g3]
// ws layout (floats): [0..127]=colsum [128..255]=colsumsq [256..383]=scale [384..511]=bias

__global__ __launch_bounds__(256) void init_out_kernel(const float4* __restrict__ emb,
                                                       float4* __restrict__ out, int nd4) {
  const int stride = gridDim.x * blockDim.x;
  const float4 z = make_float4(0.f, 0.f, 0.f, 0.f);
  for (int i = blockIdx.x * blockDim.x + threadIdx.x; i < nd4; i += stride) {
    float4 v = emb[i];
    out[i] = v;                                // lats[0] = embeds
    out[(size_t)4 * nd4 + i] = v;              // gcn_lats[0] = embeds
    out[(size_t)5 * nd4 + i] = z;              // g1 accumulator
    out[(size_t)6 * nd4 + i] = z;              // g2 accumulator
    out[(size_t)7 * nd4 + i] = z;              // g3 accumulator
  }
}

__global__ void zero_stats_kernel(float* __restrict__ ws) {
  ws[threadIdx.x] = 0.0f;                      // launched with 256 threads
}

// Column sum & sum-of-squares of x (NTOT x DIM) via WMMA:
//   D = ones(16x32,f16) x B(32x16,f16)  ->  every row of D = column sums of B.
// Each wave owns a 32-row x 32-col tile staged transposed in LDS so the
// B operand (lane n -> column n, 16 contiguous f16 rows) is a 32B aligned load.
__global__ __launch_bounds__(256) void stats_wmma_kernel(const float* __restrict__ x,
                                                         float* __restrict__ stats) {
  __shared__ __align__(32) _Float16 lt[8][32][32];   // [wave][col][row], 16 KB
  const int lane  = threadIdx.x & 31;
  const int wave  = threadIdx.x >> 5;
  const int gwave = blockIdx.x * 8 + wave;
  const int strip = gwave & 3;                 // which 32-column strip of DIM=128
  const int chunk0 = gwave >> 2;               // starting 32-row chunk
  const int step   = (gridDim.x * 8) >> 2;     // waves per strip
  const int nchunks = NTOT / 32;               // 5000 (exact)
  const int col0 = strip * 32;

  v16h ones;
#pragma unroll
  for (int i = 0; i < 16; ++i) ones[i] = (_Float16)1.0f;

  v8f s0 = {}, q0 = {}, s1 = {}, q1 = {};
  const int halfsel = (lane < 16) ? 0 : 16;    // lanes 0-15: K=0..15, 16-31: K=16..31
  const int nloc = lane & 15;

  // grid sized so every wave runs the same trip count (barriers stay uniform)
  for (int ch = chunk0; ch < nchunks; ch += step) {
    const int row0 = ch * 32;
#pragma unroll
    for (int r = 0; r < 32; ++r) {             // coalesced 128B row-segment loads
      float v = x[(size_t)(row0 + r) * DIM + col0 + lane];
      lt[wave][lane][r] = (_Float16)v;         // transpose into LDS
    }
    __syncthreads();
    v16h b0 = *reinterpret_cast<const v16h*>(&lt[wave][nloc][halfsel]);       // cols 0..15
    v16h b1 = *reinterpret_cast<const v16h*>(&lt[wave][16 + nloc][halfsel]);  // cols 16..31
    v16h b0q = b0 * b0;
    v16h b1q = b1 * b1;
    s0 = __builtin_amdgcn_wmma_f32_16x16x32_f16(false, ones, false, b0,  (short)0, s0, false, false);
    q0 = __builtin_amdgcn_wmma_f32_16x16x32_f16(false, ones, false, b0q, (short)0, q0, false, false);
    s1 = __builtin_amdgcn_wmma_f32_16x16x32_f16(false, ones, false, b1,  (short)0, s1, false, false);
    q1 = __builtin_amdgcn_wmma_f32_16x16x32_f16(false, ones, false, b1q, (short)0, q1, false, false);
    __syncthreads();
  }
  // D row M=0 lives in acc VGPR 0, lanes 0..15 (N = lane)
  if (lane < 16) {
    atomicAdd(&stats[col0 + lane],            s0[0]);
    atomicAdd(&stats[col0 + 16 + lane],       s1[0]);
    atomicAdd(&stats[DIM + col0 + lane],      q0[0]);
    atomicAdd(&stats[DIM + col0 + 16 + lane], q1[0]);
  }
}

__global__ void finalize_stats_kernel(const float* __restrict__ gamma,
                                      const float* __restrict__ beta,
                                      float* __restrict__ ws) {
  const int c = threadIdx.x;                   // 128 threads
  const float invN = 1.0f / (float)NTOT;
  float mean = ws[c] * invN;
  float var  = ws[DIM + c] * invN - mean * mean;   // population variance (ddof=0)
  float s = gamma[c] * rsqrtf(var + EPS_BN);
  ws[2 * DIM + c] = s;
  ws[3 * DIM + c] = beta[c] - mean * s;
}

__global__ __launch_bounds__(256) void bn_apply_kernel(const float4* __restrict__ x,
                                                       const float* __restrict__ ws,
                                                       float4* __restrict__ h, int nd4) {
  const float4* sc = (const float4*)(ws + 2 * DIM);
  const float4* bi = (const float4*)(ws + 3 * DIM);
  const int stride = gridDim.x * blockDim.x;
  for (int i = blockIdx.x * blockDim.x + threadIdx.x; i < nd4; i += stride) {
    const int c4 = i & 31;                     // 32 float4 per row of 128
    float4 v = x[i];
    float4 s = sc[c4];
    float4 b = bi[c4];
    float4 o;
    o.x = fmaf(v.x, s.x, b.x);
    o.y = fmaf(v.y, s.y, b.y);
    o.z = fmaf(v.z, s.z, b.z);
    o.w = fmaf(v.w, s.w, b.w);
    h[i] = o;
  }
}

// One edge per wave: lane L handles dims 4L..4L+3.  Gathers are 512B
// contiguous per wave; h and gcn slabs (82 MB each) are L2-resident on
// MI455X (192 MB L2), so the fp32 atomics stay on-chip.
__global__ __launch_bounds__(256) void scatter_kernel(const float* __restrict__ h,
                                                      const float* __restrict__ vals,
                                                      const int* __restrict__ rows,
                                                      const int* __restrict__ cols,
                                                      float* __restrict__ gcn, int nnz) {
  const int lane  = threadIdx.x & 31;
  const int gw    = blockIdx.x * (blockDim.x >> 5) + (threadIdx.x >> 5);
  const int wstep = gridDim.x * (blockDim.x >> 5);
  const float4* hu = (const float4*)h;                              // user rows
  const float4* hi = (const float4*)(h + (size_t)USER_N * DIM);     // item rows
  float* gu = gcn;
  float* gi = gcn + (size_t)USER_N * DIM;
  for (int e = gw; e < nnz; e += wstep) {
    const int   r = rows[e];
    const int   c = cols[e];
    const float v = vals[e];
    const float4 a = hi[(size_t)c * 32 + lane];   // item row segment
    const float4 b = hu[(size_t)r * 32 + lane];   // user row segment
    float* pu = gu + (size_t)r * DIM + lane * 4;
    float* pi = gi + (size_t)c * DIM + lane * 4;
    atomicAdd(pu + 0, v * a.x);
    atomicAdd(pu + 1, v * a.y);
    atomicAdd(pu + 2, v * a.z);
    atomicAdd(pu + 3, v * a.w);
    atomicAdd(pi + 0, v * b.x);
    atomicAdd(pi + 1, v * b.y);
    atomicAdd(pi + 2, v * b.z);
    atomicAdd(pi + 3, v * b.w);
  }
}

__global__ __launch_bounds__(256) void residual_kernel(const float4* __restrict__ prev,
                                                       const float4* __restrict__ g,
                                                       float4* __restrict__ nxt, int nd4) {
  const int stride = gridDim.x * blockDim.x;
  for (int i = blockIdx.x * blockDim.x + threadIdx.x; i < nd4; i += stride) {
    float4 p = prev[i];
    float4 q = g[i];
    float4 o;
    o.x = p.x + q.x;
    o.y = p.y + q.y;
    o.z = p.z + q.z;
    o.w = p.w + q.w;
    nxt[i] = o;
  }
}

extern "C" void kernel_launch(void* const* d_in, const int* in_sizes, int n_in,
                              void* d_out, int out_size, void* d_ws, size_t ws_size,
                              hipStream_t stream) {
  (void)n_in; (void)out_size; (void)ws_size;
  const float* embeds   = (const float*)d_in[0];
  const float* adj_vals = (const float*)d_in[1];
  const float* gamma    = (const float*)d_in[2];
  const float* beta     = (const float*)d_in[3];
  const int*   adj_rows = (const int*)d_in[4];
  const int*   adj_cols = (const int*)d_in[5];
  const int    nnz      = in_sizes[1];

  float* out = (float*)d_out;
  float* ws  = (float*)d_ws;
  const size_t nd  = (size_t)NTOT * DIM;   // 20,480,000
  const int    nd4 = (int)(nd / 4);

  init_out_kernel<<<2048, 256, 0, stream>>>((const float4*)embeds, (float4*)out, nd4);

  for (int i = 0; i < 3; ++i) {
    float* x = out + (size_t)i * nd;         // lats[i]
    float* h = out + (size_t)(i + 1) * nd;   // temp for batchnorm(h), becomes lats[i+1]
    float* g = out + (size_t)(5 + i) * nd;   // gcn_lats[i+1] (pre-zeroed)

    zero_stats_kernel<<<1, 256, 0, stream>>>(ws);
    stats_wmma_kernel<<<250, 256, 0, stream>>>(x, ws);   // 2000 waves, uniform 10 trips
    finalize_stats_kernel<<<1, 128, 0, stream>>>(gamma + i * DIM, beta + i * DIM, ws);
    bn_apply_kernel<<<2048, 256, 0, stream>>>((const float4*)x, ws, (float4*)h, nd4);
    scatter_kernel<<<4096, 256, 0, stream>>>(h, adj_vals, adj_rows, adj_cols, g, nnz);
    residual_kernel<<<2048, 256, 0, stream>>>((const float4*)x, (const float4*)g, (float4*)h, nd4);
  }
}
